// CGRNN_22187801051712
// MI455X (gfx1250) — matrix operand order repaired and checked by reading, not compiled, not run on hardware
//
// CDNA5 / gfx1250 implementation of the CGRNN reference.
// All contractions via v_wmma_f32_16x16x32_f16 (f16 in / f32 acc); activations
// staged through LDS with vectorized b128 global loads + b64 LDS stores. A
// FULL-tile specialization (no bounds checks) is dispatched when M/N/K are
// tile-aligned (covers the FLOP-dominant 512-wide adjacency / gating GEMMs).
// Working set (~200MB) is L2-resident on MI455X (192MB L2, 23.3TB/s HBM) so
// the kernels are matrix-pipe bound, not bandwidth bound.
#include <hip/hip_runtime.h>
#include <math.h>

typedef __attribute__((ext_vector_type(16))) _Float16 v16h;
typedef __attribute__((ext_vector_type(8)))  float    v8f;
typedef __attribute__((ext_vector_type(4)))  _Float16 v4h;

#define BM 64
#define BN 128
#define BK 32
#define LDS_A 36   // padded half-stride (32+4) to spread LDS banks
#define LDS_B 36

struct GemmP {
  const float* A; const float* B; float* C; const float* bias;
  int M, N, K;
  int lda, ldb, ldc;              // row strides (elements)
  long sA1, sA2, sB1, sB2, sC1, sC2;  // batch strides: z = z1*BH + z2
  int BH;
  float alpha;
  int biasMode;                   // 0 none, 1 per-col, 2 per-row
  int act;                        // 0 none, 1 relu
  int bT;                         // B stored (N,K) row-major (i.e. op(B)=B^T)
};

__device__ __forceinline__ v4h cvt4(const float4 f) {
  v4h h; h[0] = (_Float16)f.x; h[1] = (_Float16)f.y;
         h[2] = (_Float16)f.z; h[3] = (_Float16)f.w;
  return h;
}

// ---------------------------------------------------------------------------
// Generic batched GEMM: C = act(alpha * A(MxK) * opB(KxN) + bias)
// 256 threads = 8 waves; macro tile 64x128; each wave: 32x32 (2x2 WMMA tiles).
// FULL=true: all dims tile-aligned -> no bounds checks anywhere.
// ---------------------------------------------------------------------------
template <bool FULL>
__global__ __launch_bounds__(256, 2) void wmma_gemm_f16(GemmP p) {
  __shared__ _Float16 As[BM * LDS_A];      // [m][k]
  __shared__ _Float16 Bt[BN * LDS_B];      // [n][k] (transposed tile)

  const int z  = blockIdx.z;
  const int z1 = z / p.BH, z2 = z % p.BH;
  const float* A = p.A + z1 * p.sA1 + z2 * p.sA2;
  const float* B = p.B + z1 * p.sB1 + z2 * p.sB2;
  float*       C = p.C + z1 * p.sC1 + z2 * p.sC2;

  const int m0  = blockIdx.y * BM;
  const int n0  = blockIdx.x * BN;
  const int tid = threadIdx.x;
  const int lane = tid & 31, wave = tid >> 5;
  const int wm = (wave & 1) * 32;          // wave row offset   (2 waves in M)
  const int wn = (wave >> 1) * 32;         // wave col offset   (4 waves in N)

  // staging coordinates (computed once)
  const int ai   = tid >> 2;               // A: row 0..63
  const int akk0 = (tid & 3) * 8;          // A: k offset {0,8,16,24}
  const int bnl  = tid >> 1;               // B^T: n row 0..127
  const int bkk0 = (tid & 1) * 16;         // B^T: k offset {0,16}
  const int bkk  = tid >> 3;               // B(NN): k row 0..31
  const int bnl0 = (tid & 7) * 16;         // B(NN): n offset

  v8f acc[4];
  #pragma unroll
  for (int i = 0; i < 4; ++i)
    #pragma unroll
    for (int e = 0; e < 8; ++e) acc[i][e] = 0.0f;

  for (int k0 = 0; k0 < p.K; k0 += BK) {
    // ---- stage A tile (64x32), f32 -> f16 ----
    {
      const int gm = m0 + ai;
      const float* src = A + (long)gm * p.lda + k0 + akk0;
      if (k0 + BK < p.K) __builtin_prefetch(src + BK, 0, 1);  // uniform guard
      _Float16* dst = &As[ai * LDS_A + akk0];
      if (FULL || (gm < p.M && k0 + BK <= p.K)) {
        ((v4h*)dst)[0] = cvt4(((const float4*)src)[0]);
        ((v4h*)dst)[1] = cvt4(((const float4*)src)[1]);
      } else {
        #pragma unroll
        for (int j = 0; j < 8; ++j) {
          float v = (gm < p.M && (k0 + akk0 + j) < p.K) ? src[j] : 0.0f;
          dst[j] = (_Float16)v;
        }
      }
    }
    // ---- stage B tile into Bt[n][k] ----
    if (p.bT) {                            // B stored (N,K): rows along K
      const int gn = n0 + bnl;
      const float* src = B + (long)gn * p.ldb + k0 + bkk0;
      if (k0 + BK < p.K) __builtin_prefetch(src + BK, 0, 1);
      _Float16* dst = &Bt[bnl * LDS_B + bkk0];
      if (FULL || (gn < p.N && k0 + BK <= p.K)) {
        #pragma unroll
        for (int q = 0; q < 4; ++q)
          ((v4h*)dst)[q] = cvt4(((const float4*)src)[q]);
      } else {
        #pragma unroll
        for (int j = 0; j < 16; ++j) {
          float v = (gn < p.N && (k0 + bkk0 + j) < p.K) ? src[j] : 0.0f;
          dst[j] = (_Float16)v;
        }
      }
    } else {                               // B stored (K,N): rows along N
      const float* src = B + (long)(k0 + bkk) * p.ldb + n0 + bnl0;
      if (FULL || ((k0 + bkk) < p.K && (n0 + bnl0 + 16 <= p.N))) {
        #pragma unroll
        for (int q = 0; q < 4; ++q) {
          const float4 f = ((const float4*)src)[q];
          Bt[(bnl0 + 4 * q + 0) * LDS_B + bkk] = (_Float16)f.x;
          Bt[(bnl0 + 4 * q + 1) * LDS_B + bkk] = (_Float16)f.y;
          Bt[(bnl0 + 4 * q + 2) * LDS_B + bkk] = (_Float16)f.z;
          Bt[(bnl0 + 4 * q + 3) * LDS_B + bkk] = (_Float16)f.w;
        }
      } else {
        const bool kok = (k0 + bkk) < p.K;
        #pragma unroll
        for (int j = 0; j < 16; ++j) {
          const int nl = bnl0 + j;
          float v = (kok && (n0 + nl) < p.N) ? src[j] : 0.0f;
          Bt[nl * LDS_B + bkk] = (_Float16)v;
        }
      }
    }
    __syncthreads();

    // ---- gather fragments, contiguous K order (k = 16*half + j) for A and B ----
    const int half = lane >> 4;            // lane group 0/1
    const int l15  = lane & 15;
    const int kbA  = half * 16;            // A: K base for this lane group
    const int kbB  = half * 16;            // B: K base for this lane group
    v16h aF[2], bF[2];
    #pragma unroll
    for (int t = 0; t < 2; ++t) {
      const _Float16* ap = &As[(wm + l15 + t * 16) * LDS_A];
      #pragma unroll
      for (int j = 0; j < 16; ++j) aF[t][j] = ap[kbA + j];   // contiguous K
      const _Float16* bp = &Bt[(wn + l15 + t * 16) * LDS_B + kbB];
      #pragma unroll
      for (int j = 0; j < 16; ++j) bF[t][j] = bp[j];   // contiguous K pairs
    }
    acc[0] = __builtin_amdgcn_wmma_f32_16x16x32_f16(false, aF[0], false, bF[0], (short)0, acc[0], false, false);
    acc[1] = __builtin_amdgcn_wmma_f32_16x16x32_f16(false, aF[0], false, bF[1], (short)0, acc[1], false, false);
    acc[2] = __builtin_amdgcn_wmma_f32_16x16x32_f16(false, aF[1], false, bF[0], (short)0, acc[2], false, false);
    acc[3] = __builtin_amdgcn_wmma_f32_16x16x32_f16(false, aF[1], false, bF[1], (short)0, acc[3], false, false);
    __syncthreads();
  }

  // ---- epilogue: C layout = VGPR r -> M=r (lanes 0-15) / r+8 (lanes 16-31)
  #pragma unroll
  for (int tm = 0; tm < 2; ++tm)
    #pragma unroll
    for (int tn = 0; tn < 2; ++tn) {
      const int col = n0 + wn + tn * 16 + (lane & 15);
      const bool cok = FULL || col < p.N;
      const float bcol = (p.biasMode == 1 && cok) ? p.bias[col] : 0.0f;
      #pragma unroll
      for (int r = 0; r < 8; ++r) {
        const int row = m0 + wm + tm * 16 + r + (lane >> 4) * 8;
        if (FULL || (cok && row < p.M)) {
          float v = acc[tm * 2 + tn][r] * p.alpha + bcol;
          if (p.biasMode == 2) v += p.bias[row];
          if (p.act == 1) v = fmaxf(v, 0.0f);
          C[(long)row * p.ldc + col] = v;
        }
      }
    }
}

// ---------------------------------------------------------------------------
// Row softmax over 512-wide rows (attention), block=256.
// ---------------------------------------------------------------------------
__global__ void softmax512(float* S) {
  __shared__ float red[8], red2[8];
  float* p = S + (long)blockIdx.x * 512;
  const int tid = threadIdx.x;
  float m = -1e30f;
  for (int i = tid; i < 512; i += 256) m = fmaxf(m, p[i]);
  for (int off = 16; off; off >>= 1) m = fmaxf(m, __shfl_xor(m, off, 32));
  if ((tid & 31) == 0) red[tid >> 5] = m;
  __syncthreads();
  float mm = red[0];
  #pragma unroll
  for (int w = 1; w < 8; ++w) mm = fmaxf(mm, red[w]);
  float s = 0.0f;
  for (int i = tid; i < 512; i += 256) { float e = __expf(p[i] - mm); p[i] = e; s += e; }
  for (int off = 16; off; off >>= 1) s += __shfl_xor(s, off, 32);
  if ((tid & 31) == 0) red2[tid >> 5] = s;
  __syncthreads();
  float ss = 0.0f;
  #pragma unroll
  for (int w = 0; w < 8; ++w) ss += red2[w];
  const float inv = 1.0f / ss;
  for (int i = tid; i < 512; i += 256) p[i] *= inv;
}

// y = LayerNorm(a+b)*g + beta over rows of 96; one wave32 per row.
__global__ void add_ln96(const float* a, const float* b, const float* g,
                         const float* be, float* y, int rows) {
  const int row = blockIdx.x * 8 + (threadIdx.x >> 5);
  if (row >= rows) return;
  const int lane = threadIdx.x & 31;
  const float* pa = a + (long)row * 96;
  const float* pb = b + (long)row * 96;
  float v[3]; float s = 0.0f;
  #pragma unroll
  for (int j = 0; j < 3; ++j) { const int c = lane + 32 * j; v[j] = pa[c] + pb[c]; s += v[j]; }
  for (int off = 16; off; off >>= 1) s += __shfl_xor(s, off, 32);
  const float mu = s * (1.0f / 96.0f);
  float var = 0.0f;
  #pragma unroll
  for (int j = 0; j < 3; ++j) { const float d = v[j] - mu; var += d * d; }
  for (int off = 16; off; off >>= 1) var += __shfl_xor(var, off, 32);
  const float inv = rsqrtf(var * (1.0f / 96.0f) + 1e-5f);
  #pragma unroll
  for (int j = 0; j < 3; ++j) {
    const int c = lane + 32 * j;
    y[(long)row * 96 + c] = (v[j] - mu) * inv * g[c] + be[c];
  }
}

// inp[(n*32+b)*96+d] = h[b,d,n] + embed[n,d]*mask[b,0,n,mstep]
__global__ void build_inp_k(const float* h, const float* embed, const float* mask,
                            int mstep, float* inp) {
  const int idx = blockIdx.x * blockDim.x + threadIdx.x;
  if (idx >= 512 * 32 * 96) return;
  const int n = idx / (32 * 96);
  const int r = idx % (32 * 96);
  const int b = r / 96, d = r % 96;
  inp[idx] = h[b * 96 * 512 + d * 512 + n] + embed[n * 96 + d] * mask[b * 512 * 16 + n * 16 + mstep];
}

__device__ __forceinline__ unsigned pcg32(unsigned x) {
  x = x * 747796405u + 2891336453u;
  unsigned w = ((x >> ((x >> 28) + 4u)) ^ x) * 277803737u;
  return (w >> 22) ^ w;
}

// h = sqrt_ac[ds]*h + sqrt_omac[ds]*(mu + z*sig); mu/sig in (n*32+b, d) layout.
__global__ void diffuse_update_k(float* h, const float* mu, const float* sig,
                                 const float* dtt, const float* tab, int step, int layer) {
  const int idx = blockIdx.x * blockDim.x + threadIdx.x;
  if (idx >= 32 * 96 * 512) return;
  const int b = idx / (96 * 512);
  const int r = idx % (96 * 512);
  const int d = r / 512, n = r % 512;
  const long row = (long)(n * 32 + b) * 96 + d;
  const float m = mu[row], sg = sig[row];
  unsigned sd = (unsigned)(step * 2 + layer) * 0x9E3779B9u ^ (unsigned)idx * 0x85EBCA6Bu;
  const unsigned r1 = pcg32(sd), r2 = pcg32(sd ^ 0x68BC21EBu);
  const float u1 = ((float)r1 + 1.0f) * 2.3283064e-10f;
  const float u2 = (float)r2 * 2.3283064e-10f;
  const float zn = sqrtf(-2.0f * __logf(u1)) * __cosf(6.2831853f * u2);
  const float dt = dtt[b * 16 + step];
  int ds = (int)floorf(dt * 100.0f);
  ds = ds < 0 ? 0 : (ds > 100 ? 100 : ds);
  h[idx] = tab[ds] * h[idx] + tab[128 + ds] * (m + zn * sg);
}

// layer0 xg (c=97): ch0 = x[...,step], ch1..96 = h0. Dest stride 3c per b.
__global__ void build_xg0_k(const float* x, const float* h0, int step, float* xg) {
  const int idx = blockIdx.x * blockDim.x + threadIdx.x;
  if (idx >= 32 * 97 * 512) return;
  const int b = idx / (97 * 512);
  const int r = idx % (97 * 512);
  const int ch = r / 512, n = r % 512;
  const float v = (ch == 0) ? x[b * 512 * 16 + n * 16 + step]
                            : h0[b * 96 * 512 + (ch - 1) * 512 + n];
  xg[(long)b * (3 * 97 * 512) + ch * 512 + n] = v;
}

// layer1 xg (c=192): ch0..95 = h0(new), ch96..191 = h1. Dest stride 3c per b.
__global__ void build_xg1_k(const float* h0, const float* h1, float* xg) {
  const int idx = blockIdx.x * blockDim.x + threadIdx.x;
  if (idx >= 32 * 192 * 512) return;
  const int b = idx / (192 * 512);
  const int r = idx % (192 * 512);
  const int ch = r / 512, n = r % 512;
  const float v = (ch < 96) ? h0[b * 96 * 512 + ch * 512 + n]
                            : h1[b * 96 * 512 + (ch - 96) * 512 + n];
  xg[(long)b * (3 * 192 * 512) + ch * 512 + n] = v;
}

// h = sigmoid(u)*h + (1-sigmoid(u))*tanh(c); conditional output write.
__global__ void gate_k(const float* uc, float* h, const int* lengths,
                       int step, int writeOut, float* out) {
  const int idx = blockIdx.x * blockDim.x + threadIdx.x;
  if (idx >= 32 * 96 * 512) return;
  const int b = idx / (96 * 512);
  const int r = idx % (96 * 512);
  const int o = r / 512, n = r % 512;
  float u  = uc[(long)b * 192 * 512 + o * 512 + n];
  float cv = uc[(long)b * 192 * 512 + (96 + o) * 512 + n];
  u = 1.0f / (1.0f + __expf(-u));
  cv = tanhf(cv);
  const float hn = u * h[idx] + (1.0f - u) * cv;
  h[idx] = hn;
  if (writeOut && (lengths[b] - 1 == step)) out[idx] = hn;
}

// Collapse [x,x1,x2,x1,x2] weights: Wst = [W0 | W1+W3 | W2+W4] (192 x 3c),
// stored with row stride ldw (padded for 16B-aligned rows).
__global__ void build_wstack_k(const float* uw, const float* cw, const float* ub,
                               const float* cb, float* wst, float* bst,
                               int c, int ldw) {
  const int idx = blockIdx.x * blockDim.x + threadIdx.x;
  if (idx < 192) bst[idx] = (idx < 96) ? ub[idx] : cb[idx - 96];
  const int c3 = 3 * c, c5 = 5 * c;
  if (idx >= 192 * c3) return;
  const int rr = idx / c3, j = idx % c3;
  const float* w = (rr < 96) ? uw : cw;
  const int o = (rr < 96) ? rr : rr - 96;
  float v;
  if (j < c)            v = w[o * c5 + j];
  else if (j < 2 * c)   v = w[o * c5 + c + (j - c)]     + w[o * c5 + 3 * c + (j - c)];
  else                  v = w[o * c5 + 2 * c + (j - 2 * c)] + w[o * c5 + 4 * c + (j - 2 * c)];
  wst[(long)rr * ldw + j] = v;
}

__global__ void make_tab_k(float* tab) {
  if (threadIdx.x == 0 && blockIdx.x == 0) {
    const float bend = sqrtf(2e-5f), bstart = sqrtf(1e-5f);
    float ac = 1.0f;
    for (int i = 0; i <= 100; ++i) {
      const float beta = bend + (bstart - bend) * (float)i / 100.0f;
      ac *= (1.0f - beta);
      tab[i] = sqrtf(ac);
      tab[128 + i] = sqrtf(fmaxf(1.0f - ac, 0.0f));
    }
  }
}

__global__ void fill_k(float* p, long n, float v) {
  const long i = (long)blockIdx.x * blockDim.x + threadIdx.x;
  if (i < n) p[i] = v;
}

// ---------------------------------------------------------------------------
static void gemm(hipStream_t s, const float* A, const float* B, float* C,
                 const float* bias, int M, int N, int K, int lda, int ldb, int ldc,
                 long sA1, long sA2, long sB1, long sB2, long sC1, long sC2,
                 int BH, int Z, float alpha, int biasMode, int act, int bT) {
  GemmP p{A, B, C, bias, M, N, K, lda, ldb, ldc, sA1, sA2, sB1, sB2, sC1, sC2,
          BH, alpha, biasMode, act, bT};
  dim3 g((N + BN - 1) / BN, (M + BM - 1) / BM, Z);
  if ((M % BM) == 0 && (N % BN) == 0 && (K % BK) == 0)
    wmma_gemm_f16<true><<<g, 256, 0, s>>>(p);    // guard-free hot path
  else
    wmma_gemm_f16<false><<<g, 256, 0, s>>>(p);
}
static inline int eb(long n) { return (int)((n + 255) / 256); }

// param leaf indices (dict insertion order from _layer_params)
enum { E_EMBED=0,E_QKVW,E_QKVB,E_OUTW,E_OUTB,E_L1W,E_L1B,E_L2W,E_L2B,
       E_LN1G,E_LN1B,E_LN2G,E_LN2B,E_MUW,E_MUB,E_SIGW,E_SIGB,E_UW,E_UB,E_CW,E_CB };

extern "C" void kernel_launch(void* const* d_in, const int* in_sizes, int n_in,
                              void* d_out, int out_size, void* d_ws, size_t ws_size,
                              hipStream_t stream) {
  if (n_in < 48) return;
  const float* x    = (const float*)d_in[0];   // (32,1,512,16)
  const float* adj  = (const float*)d_in[1];   // (512,512)
  const float* dtt  = (const float*)d_in[2];   // (32,16)
  const int*   len  = (const int*)  d_in[3];
  const float* mask = (const float*)d_in[5];   // (32,1,512,16)
  const float* P[2][21];
  for (int i = 0; i < 21; ++i) { P[0][i] = (const float*)d_in[6 + i];
                                 P[1][i] = (const float*)d_in[27 + i]; }
  float* out = (float*)d_out;
  float* W = (float*)d_ws;

  // workspace arena (floats)
  long off = 0;
  auto al = [&](long n) { long o = off; off += (n + 63) & ~63L; return o; };
  const long HSZ = 32L * 96 * 512;               // 1,572,864
  float* tab  = W + al(256);
  float* h0   = W + al(HSZ);
  float* h1   = W + al(HSZ);
  float* inp  = W + al(HSZ);
  float* qkv  = W + al(16384L * 288);
  float* sc   = W + al(64L * 512 * 512);
  float* obuf = W + al(HSZ);
  float* pbuf = W + al(HSZ);
  float* h1b  = W + al(HSZ);
  float* ffb  = W + al(HSZ);
  float* ff2  = W + al(HSZ);
  float* tb   = W + al(HSZ);
  float* mub  = W + al(HSZ);
  float* sgb  = W + al(HSZ);
  float* xg3  = W + al(32L * 576 * 512);         // sized for layer1 (3c=576)
  float* uc   = W + al(32L * 192 * 512);
  const int LDW0 = 292, LDW1 = 576;              // padded (16B-aligned) rows
  float* ws0  = W + al(192L * LDW0);
  float* ws1  = W + al(192L * LDW1);
  float* bb0  = W + al(192);
  float* bb1  = W + al(192);
  (void)ws_size; (void)in_sizes; (void)out_size;

  fill_k<<<eb(HSZ), 256, 0, stream>>>(h0, HSZ, 0.0f);
  fill_k<<<eb(HSZ), 256, 0, stream>>>(h1, HSZ, 0.0f);
  make_tab_k<<<1, 1, 0, stream>>>(tab);
  build_wstack_k<<<eb(192L * 3 * 97), 256, 0, stream>>>(P[0][E_UW], P[0][E_CW], P[0][E_UB], P[0][E_CB], ws0, bb0, 97, LDW0);
  build_wstack_k<<<eb(192L * 3 * 192), 256, 0, stream>>>(P[1][E_UW], P[1][E_CW], P[1][E_UB], P[1][E_CB], ws1, bb1, 192, LDW1);

  const float iskd = 0.14433756729740646f;       // 1/sqrt(48)
  for (int step = 0; step < 16; ++step) {
    for (int l = 0; l < 2; ++l) {
      float* h = l ? h1 : h0;
      const float** p = P[l];
      if (step != 0) {
        const int mstep = step - 1;
        // ---- transformer diffuse ----
        build_inp_k<<<eb(HSZ), 256, 0, stream>>>(h, p[E_EMBED], mask, mstep, inp);
        // qkv: (16384,96) x (288,96)^T
        gemm(stream, inp, p[E_QKVW], qkv, p[E_QKVB], 16384, 288, 96, 96, 96, 288,
             0,0,0,0,0,0, 1, 1, 1.0f, 1, 0, 1);
        // scores: per (b,h): q(512,48) x k(512,48)^T / sqrt(48)
        gemm(stream, qkv, qkv + 96, sc, nullptr, 512, 512, 48, 9216, 9216, 512,
             288, 48, 288, 48, 2L*262144, 262144, 2, 64, iskd, 0, 0, 1);
        softmax512<<<64 * 512, 256, 0, stream>>>(sc);
        // o: attn(512,512) x v(512,48)
        gemm(stream, sc, qkv + 192, obuf, nullptr, 512, 48, 512, 512, 9216, 3072,
             2L*262144, 262144, 288, 48, 96, 48, 2, 64, 1.0f, 0, 0, 0);
        // out proj, residual LN1
        gemm(stream, obuf, p[E_OUTW], pbuf, p[E_OUTB], 16384, 96, 96, 96, 96, 96,
             0,0,0,0,0,0, 1, 1, 1.0f, 1, 0, 1);
        add_ln96<<<(16384 + 7) / 8, 256, 0, stream>>>(inp, pbuf, p[E_LN1G], p[E_LN1B], h1b, 16384);
        // FFN (relu fused) + residual LN2
        gemm(stream, h1b, p[E_L1W], ffb, p[E_L1B], 16384, 96, 96, 96, 96, 96,
             0,0,0,0,0,0, 1, 1, 1.0f, 1, 1, 1);
        gemm(stream, ffb, p[E_L2W], ff2, p[E_L2B], 16384, 96, 96, 96, 96, 96,
             0,0,0,0,0,0, 1, 1, 1.0f, 1, 0, 1);
        add_ln96<<<(16384 + 7) / 8, 256, 0, stream>>>(h1b, ff2, p[E_LN2G], p[E_LN2B], tb, 16384);
        // mu / sigma heads
        gemm(stream, tb, p[E_MUW], mub, p[E_MUB], 16384, 96, 96, 96, 96, 96,
             0,0,0,0,0,0, 1, 1, 1.0f, 1, 0, 1);
        gemm(stream, tb, p[E_SIGW], sgb, p[E_SIGB], 16384, 96, 96, 96, 96, 96,
             0,0,0,0,0,0, 1, 1, 1.0f, 1, 0, 1);
        diffuse_update_k<<<eb(HSZ), 256, 0, stream>>>(h, mub, sgb, dtt, tab, step, l);
      }
      // ---- spatial conv + gated update ----
      const int c = l ? 192 : 97;
      const long bs = 3L * c * 512;              // per-b stride in xg3
      if (l == 0) build_xg0_k<<<eb(32L * 97 * 512), 256, 0, stream>>>(x, h0, step, xg3);
      else        build_xg1_k<<<eb(32L * 192 * 512), 256, 0, stream>>>(h0, h1, xg3);
      // x1 = xg @ adj^T  (per b, rows = channels); layer1 hits FULL path
      gemm(stream, xg3, adj, xg3 + (long)c * 512, nullptr, c, 512, 512, 512, 512, 512,
           bs, 0, 0, 0, bs, 0, 1, 32, 1.0f, 0, 0, 1);
      // x2 = x1 @ adj^T
      gemm(stream, xg3 + (long)c * 512, adj, xg3 + 2L * c * 512, nullptr, c, 512, 512, 512, 512, 512,
           bs, 0, 0, 0, bs, 0, 1, 32, 1.0f, 0, 0, 1);
      // [u;c] = Wstack(192,3c) @ [xg;x1;x2](3c,512) + bias-per-row
      gemm(stream, l ? ws1 : ws0, xg3, uc, l ? bb1 : bb0, 192, 512, 3 * c, l ? LDW1 : LDW0, 512, 512,
           0, 0, bs, 0, 192L * 512, 0, 1, 32, 1.0f, 2, 0, 0);
      gate_k<<<eb(HSZ), 256, 0, stream>>>(uc, h, len, step, (l == 1) ? 1 : 0, out);
    }
  }
}
